// LS_GCN_31619549233342
// MI455X (gfx1250) — matrix operand order repaired and checked by reading, not compiled, not run on hardware
//
#include <hip/hip_runtime.h>
#include <hip/hip_bf16.h>

// Problem constants (match reference)
#define NN   100000
#define FF   8
#define TT   5
#define HH   16
#define DEGC 16
#define EE   (NN * DEGC)

typedef __attribute__((ext_vector_type(16))) __bf16 v16bf;
typedef __attribute__((ext_vector_type(8)))  float  v8f;

union BFrag { v16bf v; unsigned u[8]; };

__device__ __forceinline__ unsigned short f2bf(float x) {
    unsigned u = __float_as_uint(x);
    u += 0x7FFFu + ((u >> 16) & 1u);          // round-to-nearest-even
    return (unsigned short)(u >> 16);
}
__device__ __forceinline__ unsigned pack2(float lo, float hi) {
    return (unsigned)f2bf(lo) | ((unsigned)f2bf(hi) << 16);
}

// ---------------- feature row-normalization ----------------
__global__ void k_rownorm(const float* __restrict__ feat, float* __restrict__ featn) {
    int i = blockIdx.x * blockDim.x + threadIdx.x;
    if (i >= NN) return;
    float v[FF]; float s = 0.f;
    #pragma unroll
    for (int f = 0; f < FF; f++) { v[f] = feat[i * FF + f]; s += v[f]; }
    float inv = 1.0f / s;
    #pragma unroll
    for (int f = 0; f < FF; f++) featn[i * FF + f] = v[f] * inv;
}

// ---- LSTM over sliding window; emit concat [featn | h | 0-pad] as packed bf16 pairs ----
// Output row = 16 dwords (32 bf16 cols: 0-7 featn, 8-23 h, 24-31 zero padding).
__global__ void k_lstm(const float* __restrict__ featn,
                       const float* __restrict__ W_ih, const float* __restrict__ W_hh,
                       const float* __restrict__ b_ih, const float* __restrict__ b_hh,
                       unsigned* __restrict__ hp) {
    __shared__ float sWi[64 * FF];
    __shared__ float sWh[64 * HH];
    __shared__ float sB[64];
    for (int t = threadIdx.x; t < 64 * FF; t += blockDim.x) sWi[t] = W_ih[t];
    for (int t = threadIdx.x; t < 64 * HH; t += blockDim.x) sWh[t] = W_hh[t];
    for (int t = threadIdx.x; t < 64; t += blockDim.x) sB[t] = b_ih[t] + b_hh[t];
    __syncthreads();

    int i = blockIdx.x * blockDim.x + threadIdx.x;
    if (i >= NN) return;

    float h[HH], c[HH];
    #pragma unroll
    for (int j = 0; j < HH; j++) { h[j] = 0.f; c[j] = 0.f; }

    int start = i - TT; if (start < 0) start = 0;
    int nvalid = (i < TT) ? i : TT;     // right zero-padding like pad_sequence

    for (int t = 0; t < TT; t++) {
        float x[FF];
        if (t < nvalid) {
            const float* r = &featn[(start + t) * FF];
            #pragma unroll
            for (int f = 0; f < FF; f++) x[f] = r[f];
        } else {
            #pragma unroll
            for (int f = 0; f < FF; f++) x[f] = 0.f;
        }
        float hn[HH];
        #pragma unroll
        for (int j = 0; j < HH; j++) {
            float gi = sB[j], gf = sB[j + 16], gg = sB[j + 32], go = sB[j + 48];
            #pragma unroll
            for (int f = 0; f < FF; f++) {
                float xv = x[f];
                gi += sWi[j * FF + f] * xv;
                gf += sWi[(j + 16) * FF + f] * xv;
                gg += sWi[(j + 32) * FF + f] * xv;
                go += sWi[(j + 48) * FF + f] * xv;
            }
            #pragma unroll
            for (int k = 0; k < HH; k++) {
                float hv = h[k];
                gi += sWh[j * HH + k] * hv;
                gf += sWh[(j + 16) * HH + k] * hv;
                gg += sWh[(j + 32) * HH + k] * hv;
                go += sWh[(j + 48) * HH + k] * hv;
            }
            gi = 1.f / (1.f + __expf(-gi));
            gf = 1.f / (1.f + __expf(-gf));
            go = 1.f / (1.f + __expf(-go));
            gg = tanhf(gg);
            c[j] = gf * c[j] + gi * gg;
            hn[j] = go * tanhf(c[j]);
        }
        #pragma unroll
        for (int j = 0; j < HH; j++) h[j] = hn[j];
    }

    float col[32];
    #pragma unroll
    for (int f = 0; f < FF; f++) col[f] = featn[i * FF + f];
    #pragma unroll
    for (int j = 0; j < HH; j++) col[FF + j] = h[j];
    #pragma unroll
    for (int k = 24; k < 32; k++) col[k] = 0.f;   // K padding to 32
    unsigned* op = &hp[(size_t)i * 16];
    #pragma unroll
    for (int d = 0; d < 16; d++) op[d] = pack2(col[2 * d], col[2 * d + 1]);
}

// ---------------- degree + edge normalization ----------------
__global__ void k_zero(float* p, int n) {
    int i = blockIdx.x * blockDim.x + threadIdx.x;
    if (i < n) p[i] = 0.f;
}
__global__ void k_deg(const int* __restrict__ src, const float* __restrict__ ew,
                      float* __restrict__ deg) {
    int e = blockIdx.x * blockDim.x + threadIdx.x;
    if (e >= EE) return;
    atomicAdd(&deg[src[e]], ew[e]);
}
__global__ void k_normw(const int* __restrict__ src, const int* __restrict__ dst,
                        const float* __restrict__ ew, const float* __restrict__ deg,
                        float* __restrict__ nw) {
    int e = blockIdx.x * blockDim.x + threadIdx.x;
    if (e >= EE) return;
    nw[e] = ew[e] * rsqrtf(deg[src[e]] * deg[dst[e]]);
}

// ---------------- weight repack fp32 -> bf16 WMMA B-fragments ----------------
// Layout (16-bit B 32x16, matches A): lanes 0-15: N=lane, K=0..7 & 16..23 packed 2/dword in v0..v7;
// lanes 16-31: N=lane-16, K=8..15 & 24..31.  Fragment dword t = (((kb*NT)+nt)*32+lane)*8+j.
__global__ void k_pack(const float* __restrict__ W, unsigned* __restrict__ out,
                       int Kd, int Od, int ndw) {
    int t = blockIdx.x * blockDim.x + threadIdx.x;
    if (t >= ndw) return;
    int j = t & 7;
    int lane = (t >> 3) & 31;
    int rest = t >> 8;
    int ODt = Od >> 4;
    int nt = rest % ODt;
    int kb = rest / ODt;
    int n = nt * 16 + (lane & 15);
    int k0 = kb * 32 + ((lane >> 4) * 8) + ((j >> 2) * 16) + 2 * (j & 3);
    float lo = (k0     < Kd) ? W[k0       * Od + n] : 0.f;
    float hi = (k0 + 1 < Kd) ? W[(k0 + 1) * Od + n] : 0.f;
    out[t] = pack2(lo, hi);
}

// ---------------- WMMA GEMM: M[N x Od] = Xbf16[N x 2*KPD] @ W ----------------
// Xp: packed-bf16 activations, KPD dwords per row (pairs along K, zero-padded).
// A-fragment (16-bit A 16x32): lanes 0-15 -> K 0-7 & 16-23; lanes 16-31 -> K 8-15 & 24-31.
template <int KPD, int Od>
__global__ void k_gemm(const unsigned* __restrict__ Xp, const unsigned* __restrict__ Wp,
                       float* __restrict__ M) {
    int wave = (int)((blockIdx.x * blockDim.x + threadIdx.x) >> 5);
    int lane = threadIdx.x & 31;
    int rowbase = wave * 16;
    if (rowbase >= NN) return;              // wave-uniform: EXEC stays all-1s for WMMA

    constexpr int NT = Od / 16;
    constexpr int KB = KPD / 16;            // 32 K-elements (16 dwords) per block
    const v8f vzero = {0.f, 0.f, 0.f, 0.f, 0.f, 0.f, 0.f, 0.f};
    v8f acc[NT];
    #pragma unroll
    for (int nt = 0; nt < NT; nt++) acc[nt] = vzero;

    // Per-lane A base: row = rowbase+(lane&15); K-half offset in dwords = (lane>>4)*4
    const unsigned* xrow = Xp + (size_t)(rowbase + (lane & 15)) * KPD + (lane >> 4) * 4;

    #pragma unroll
    for (int kb = 0; kb < KB; kb++) {
        BFrag a;
        const uint4 p0 = *(const uint4*)(xrow + kb * 16);      // K pairs 0..7 of half
        const uint4 p1 = *(const uint4*)(xrow + kb * 16 + 8);  // K pairs +16
        a.u[0] = p0.x; a.u[1] = p0.y; a.u[2] = p0.z; a.u[3] = p0.w;
        a.u[4] = p1.x; a.u[5] = p1.y; a.u[6] = p1.z; a.u[7] = p1.w;
        #pragma unroll
        for (int nt = 0; nt < NT; nt++) {
            BFrag b;
            const uint4* bp = (const uint4*)&Wp[(((kb * NT) + nt) * 32 + lane) * 8];
            const uint4 q0 = bp[0]; const uint4 q1 = bp[1];
            b.u[0] = q0.x; b.u[1] = q0.y; b.u[2] = q0.z; b.u[3] = q0.w;
            b.u[4] = q1.x; b.u[5] = q1.y; b.u[6] = q1.z; b.u[7] = q1.w;
            acc[nt] = __builtin_amdgcn_wmma_f32_16x16x32_bf16(
                false, a.v, false, b.v, (short)0, acc[nt], false, false);
        }
    }
    // C/D layout: lane gives column n=lane&15; rows m=(lane>>4)*8 + r
    int mo = (lane >> 4) * 8;
    int n = lane & 15;
    #pragma unroll
    for (int nt = 0; nt < NT; nt++)
        #pragma unroll
        for (int r = 0; r < 8; r++)
            M[(size_t)(rowbase + mo + r) * Od + nt * 16 + n] = acc[nt][r];
}

// ---------------- agg = bias (broadcast) ----------------
template <int Od>
__global__ void k_init_bias(const float* __restrict__ b, float* __restrict__ agg) {
    int t = blockIdx.x * blockDim.x + threadIdx.x;
    if (t >= NN * Od) return;
    agg[t] = b[t % Od];
}

// ---------------- edge scatter: agg[dst] += nw * M[src] ----------------
template <int Od>
__global__ void k_scatter(const float* __restrict__ M, const int* __restrict__ src,
                          const int* __restrict__ dst, const float* __restrict__ nw,
                          float* __restrict__ agg) {
    const int GP = Od / 4;
    int t = blockIdx.x * blockDim.x + threadIdx.x;
    if (t >= EE * GP) return;
    int e = t / GP;
    int fo = (t % GP) * 4;
    float w = nw[e];
    const float4 v = *(const float4*)&M[(size_t)src[e] * Od + fo];
    float* ap = &agg[(size_t)dst[e] * Od + fo];
    atomicAdd(ap + 0, w * v.x);
    atomicAdd(ap + 1, w * v.y);
    atomicAdd(ap + 2, w * v.z);
    atomicAdd(ap + 3, w * v.w);
}

// ------- leaky relu fused with bf16 pair packing (agg[Nx64] -> hp[Nx64 bf16]) -------
__global__ void k_leakypack(const float* __restrict__ agg, unsigned* __restrict__ hp) {
    int t = blockIdx.x * blockDim.x + threadIdx.x;   // one packed dword per thread
    if (t >= NN * 32) return;
    float x0 = agg[2 * t];
    float x1 = agg[2 * t + 1];
    x0 = (x0 > 0.f) ? x0 : 0.01f * x0;
    x1 = (x1 > 0.f) ? x1 : 0.01f * x1;
    hp[t] = pack2(x0, x1);
}

static inline int cdiv(long long a, long long b) { return (int)((a + b - 1) / b); }

extern "C" void kernel_launch(void* const* d_in, const int* in_sizes, int n_in,
                              void* d_out, int out_size, void* d_ws, size_t ws_size,
                              hipStream_t stream) {
    const float* feat  = (const float*)d_in[0];
    const int*   esrc  = (const int*)d_in[1];
    const int*   edst  = (const int*)d_in[2];
    const float* ew    = (const float*)d_in[3];
    const float* W_ih  = (const float*)d_in[4];
    const float* W_hh  = (const float*)d_in[5];
    const float* b_ih  = (const float*)d_in[6];
    const float* b_hh  = (const float*)d_in[7];
    const float* W0 = (const float*)d_in[8];  const float* b0 = (const float*)d_in[9];
    const float* W1 = (const float*)d_in[10]; const float* b1 = (const float*)d_in[11];
    const float* W2 = (const float*)d_in[12]; const float* b2 = (const float*)d_in[13];
    float* out = (float*)d_out;

    // Workspace carve-up (all 16B-aligned): ~75 MB
    float* ws = (float*)d_ws;
    size_t o = 0;
    float* featn = ws + o; o += (size_t)NN * 8;
    unsigned* hp = (unsigned*)(ws + o); o += (size_t)NN * 32;  // packed-bf16 activations
    float* mbuf  = ws + o; o += (size_t)NN * 64;               // GEMM output
    float* agg   = ws + o; o += (size_t)NN * 64;               // scatter accumulator
    float* deg   = ws + o; o += (size_t)NN;
    float* nw    = ws + o; o += (size_t)EE;
    unsigned* wp0 = (unsigned*)(ws + o);           // 1024 dwords (Kp=32, Od=64)
    unsigned* wp1 = wp0 + 1024;                    // 2048 dwords (Kp=64, Od=64)
    unsigned* wp2 = wp1 + 2048;                    // 1024 dwords (Kp=64, Od=32)

    const int B = 256;

    // 1) normalize + LSTM + concat (packed bf16, K padded 24->32)
    k_rownorm<<<cdiv(NN, B), B, 0, stream>>>(feat, featn);
    k_lstm<<<cdiv(NN, B), B, 0, stream>>>(featn, W_ih, W_hh, b_ih, b_hh, hp);

    // 2) degree + symmetric edge norm
    k_zero<<<cdiv(NN, B), B, 0, stream>>>(deg, NN);
    k_deg<<<cdiv(EE, B), B, 0, stream>>>(esrc, ew, deg);
    k_normw<<<cdiv(EE, B), B, 0, stream>>>(esrc, edst, ew, deg, nw);

    // 3) repack weights to bf16 WMMA fragments
    k_pack<<<cdiv(1024, B), B, 0, stream>>>(W0, wp0, 24, 64, 1024);
    k_pack<<<cdiv(2048, B), B, 0, stream>>>(W1, wp1, 64, 64, 2048);
    k_pack<<<cdiv(1024, B), B, 0, stream>>>(W2, wp2, 64, 32, 1024);

    const int gemm_threads = (NN / 16) * 32;       // 6250 waves, exact M tiling

    // Layer 0: [N x 24(->32)] -> [N x 64], leaky relu
    k_gemm<16, 64><<<cdiv(gemm_threads, B), B, 0, stream>>>(hp, wp0, mbuf);
    k_init_bias<64><<<cdiv((long long)NN * 64, B), B, 0, stream>>>(b0, agg);
    k_scatter<64><<<cdiv((long long)EE * 16, B), B, 0, stream>>>(mbuf, esrc, edst, nw, agg);
    k_leakypack<<<cdiv((long long)NN * 32, B), B, 0, stream>>>(agg, hp);

    // Layer 1: [N x 64] -> [N x 64], leaky relu
    k_gemm<32, 64><<<cdiv(gemm_threads, B), B, 0, stream>>>(hp, wp1, mbuf);
    k_init_bias<64><<<cdiv((long long)NN * 64, B), B, 0, stream>>>(b1, agg);
    k_scatter<64><<<cdiv((long long)EE * 16, B), B, 0, stream>>>(mbuf, esrc, edst, nw, agg);
    k_leakypack<<<cdiv((long long)NN * 32, B), B, 0, stream>>>(agg, hp);

    // Layer 2: [N x 64] -> [N x 32], no activation; scatter straight into d_out
    k_gemm<32, 32><<<cdiv(gemm_threads, B), B, 0, stream>>>(hp, wp2, mbuf);
    k_init_bias<32><<<cdiv((long long)NN * 32, B), B, 0, stream>>>(b2, out);
    k_scatter<32><<<cdiv((long long)EE * 8, B), B, 0, stream>>>(mbuf, esrc, edst, nw, out);
}
